// BertSelfAttention_44040594653775
// MI455X (gfx1250) — matrix operand order repaired and checked by reading, not compiled, not run on hardware
//
#include <hip/hip_runtime.h>

// ---- problem constants (B=2, S=2048, D=1024, H=16, DK=64) ----
#define BB 2
#define SS 2048
#define DD 1024
#define HH 16
#define DKK 64

typedef _Float16 h8  __attribute__((ext_vector_type(8)));
typedef _Float16 h16 __attribute__((ext_vector_type(16)));
typedef float    f8  __attribute__((ext_vector_type(8)));

__device__ __forceinline__ h16 cat8(h8 lo, h8 hi) {
  return __builtin_shufflevector(lo, hi, 0,1,2,3,4,5,6,7,8,9,10,11,12,13,14,15);
}

// D(16x16,f32) = A(16x32,f16) x B(32x16,f16) + C
__device__ __forceinline__ f8 wmma_f16(h16 a, h16 b, f8 c) {
  return __builtin_amdgcn_wmma_f32_16x16x32_f16(false, a, false, b, (short)0, c,
                                                false, false);
}

// CDNA5 async global->LDS copy (ASYNCcnt path). LDS aperture: generic addr
// low 32 bits == LDS byte offset (ISA 10.2), so truncation is valid.
__device__ __forceinline__ void async_copy_b128(void* lds_dst, const void* gsrc) {
  unsigned int loff = (unsigned int)(unsigned long long)(size_t)lds_dst;
  unsigned long long ga = (unsigned long long)(size_t)gsrc;
  asm volatile("global_load_async_to_lds_b128 %0, %1, off"
               :: "v"(loff), "v"(ga) : "memory");
}
__device__ __forceinline__ void wait_async0() {
  asm volatile("s_wait_asynccnt 0" ::: "memory");
}

// -------------------------------------------------------------------------
// Kernel 1: fp32 -> fp16 conversion (grid-stride, memory bound, tiny cost)
// -------------------------------------------------------------------------
__global__ void f32_to_f16_kernel(const float* __restrict__ src,
                                  _Float16* __restrict__ dst, int n) {
  int i = blockIdx.x * blockDim.x + threadIdx.x;
  int stride = gridDim.x * blockDim.x;
  for (; i < n; i += stride) dst[i] = (_Float16)src[i];
}

// -------------------------------------------------------------------------
// Kernel 2: projection GEMM  Y[m,n] = sum_k X[m,k]*W[n,k] + bias[n]
//   M=4096 (B*S), N=1024, K=1024. Block tile 128x64, 8 waves (wave32),
//   each wave: 16 m-rows x 64 n-cols = 4 accumulators.
//   mode 0: Q -> scale by 0.125, store [B,H,S,DK] f16
//   mode 1: K -> store [B,H,S,DK] f16
//   mode 2: V -> store transposed [B,H,DK,S] f16
// -------------------------------------------------------------------------
#define XSTR 40  // 32 halves + 8 pad (80B rows, 16B aligned)
#define WSTR 40

__global__ __launch_bounds__(256)
void proj_kernel(const _Float16* __restrict__ X, const _Float16* __restrict__ W,
                 const float* __restrict__ bias, _Float16* __restrict__ Y,
                 int mode) {
  __shared__ _Float16 XL[128 * XSTR];
  __shared__ _Float16 WL[64 * WSTR];

  const int tid  = threadIdx.x;
  const int lane = tid & 31;
  const int wv   = tid >> 5;
  const int lm   = lane & 15;
  const int half = lane >> 4;
  const int mBase = blockIdx.x * 128;
  const int nBase = blockIdx.y * 64;

  f8 acc[4] = {};

  for (int kc = 0; kc < DD; kc += 32) {
    // stage X tile (128 rows x 64B) and W tile (64 rows x 64B) via async DMA
#pragma unroll
    for (int p = 0; p < 2; ++p) {
      int idx = p * 256 + tid;            // 512 x 16B segments
      int row = idx >> 2, s16 = idx & 3;
      async_copy_b128(&XL[row * XSTR + s16 * 8],
                      X + (size_t)(mBase + row) * DD + kc + s16 * 8);
    }
    {
      int row = tid >> 2, s16 = tid & 3;  // 256 x 16B segments
      async_copy_b128(&WL[row * WSTR + s16 * 8],
                      W + (size_t)(nBase + row) * DD + kc + s16 * 8);
    }
    wait_async0();
    __syncthreads();

    // A fragment: row = wv*16 + lm ; K layout e -> e + 16*half (contiguous)
    const _Float16* xr = &XL[(wv * 16 + lm) * XSTR];
    h16 a = cat8(*(const h8*)(xr + 16 * half), *(const h8*)(xr + 16 * half + 8));
#pragma unroll
    for (int nt = 0; nt < 4; ++nt) {
      // B fragment: lane col n = nt*16+lm, K = e + 16*half (contiguous)
      const _Float16* wr = &WL[(nt * 16 + lm) * WSTR + 16 * half];
      h16 b = cat8(*(const h8*)wr, *(const h8*)(wr + 8));
      acc[nt] = wmma_f16(a, b, acc[nt]);
    }
    __syncthreads();
  }

  // epilogue: bias (+Q scale), convert to f16, scatter per-head
#pragma unroll
  for (int nt = 0; nt < 4; ++nt) {
    int n  = nBase + nt * 16 + lm;
    int hh = n >> 6, dd = n & 63;
    float bn = bias[n];
#pragma unroll
    for (int r = 0; r < 8; ++r) {
      int m  = mBase + wv * 16 + r + 8 * half;   // C layout row
      int bb = m >> 11, ss = m & (SS - 1);
      float val = acc[nt][r] + bn;
      if (mode == 0) val *= 0.125f;              // 1/sqrt(DK) folded into Q
      size_t o = (mode == 2)
                     ? ((size_t)((bb * HH + hh) * DKK + dd) * SS + ss)   // V^T
                     : ((size_t)((bb * HH + hh) * SS + ss) * DKK + dd);  // Q,K
      Y[o] = (_Float16)val;
    }
  }
}

// -------------------------------------------------------------------------
// Kernel 3: flash attention. Grid (S/64, H, B), 128 threads = 4 waves.
// Each wave: 16 q-rows. Stream K/V in 64-wide chunks through LDS (async),
// online softmax with shfl_xor row reductions, O accumulated via WMMA.
// -------------------------------------------------------------------------
#define KVSTR 72  // 64 halves + 8 pad (144B rows, 16B aligned)

__global__ __launch_bounds__(128)
void attn_kernel(const _Float16* __restrict__ Qh, const _Float16* __restrict__ Kh,
                 const _Float16* __restrict__ Vt, const unsigned char* __restrict__ msk,
                 float* __restrict__ Out) {
  __shared__ _Float16 KL[64 * KVSTR];        // K chunk: rows=k-local, cols=d
  __shared__ _Float16 VL[64 * KVSTR];        // V^T chunk: rows=d, cols=k-local
  __shared__ _Float16 PL[4 * 16 * KVSTR];    // per-wave P staging (C->A relayout)

  const int tid  = threadIdx.x;
  const int lane = tid & 31;
  const int wv   = tid >> 5;
  const int lm   = lane & 15;
  const int half = lane >> 4;
  const int b = blockIdx.z, h = blockIdx.y, q0 = blockIdx.x * 64;
  const size_t bh = (size_t)(b * HH + h);

  const _Float16* Qb = Qh + bh * SS * DKK;
  const _Float16* Kb = Kh + bh * SS * DKK;
  const _Float16* Vb = Vt + bh * DKK * SS;

  // Q fragments for this wave's 16 rows (row-major [S,DK], contiguous loads)
  const _Float16* qr = Qb + (size_t)(q0 + wv * 16 + lm) * DKK;
  h16 qa0 = cat8(*(const h8*)(qr + 16 * half),      *(const h8*)(qr + 16 * half + 8));
  h16 qa1 = cat8(*(const h8*)(qr + 32 + 16 * half), *(const h8*)(qr + 32 + 16 * half + 8));

  f8 o[4] = {};
  float mrow[8], lrow[8];
#pragma unroll
  for (int r = 0; r < 8; ++r) { mrow[r] = -3.0e38f; lrow[r] = 0.0f; }

  for (int kc = 0; kc < SS; kc += 64) {
    // stage K (64 x 128B rows) and V^T (64 x 128B rows) chunks
#pragma unroll
    for (int p = 0; p < 4; ++p) {
      int idx = p * 128 + tid;               // 512 x 16B segments
      int row = idx >> 3, s16 = idx & 7;
      async_copy_b128(&KL[row * KVSTR + s16 * 8],
                      Kb + (size_t)(kc + row) * DKK + s16 * 8);
    }
#pragma unroll
    for (int p = 0; p < 4; ++p) {
      int idx = p * 128 + tid;
      int row = idx >> 3, s16 = idx & 7;
      async_copy_b128(&VL[row * KVSTR + s16 * 8],
                      Vb + (size_t)row * SS + kc + s16 * 8);
    }
    wait_async0();
    __syncthreads();

    // scores: 4 tiles of 16(q) x 16(k), each = 2 WMMAs over d
    f8 st[4];
#pragma unroll
    for (int n = 0; n < 4; ++n) {
      const _Float16* kr = &KL[(n * 16 + lm) * KVSTR];
      h16 b0 = cat8(*(const h8*)(kr + 16 * half),      *(const h8*)(kr + 16 * half + 8));
      h16 b1 = cat8(*(const h8*)(kr + 32 + 16 * half), *(const h8*)(kr + 32 + 16 * half + 8));
      f8 s = {};
      s = wmma_f16(qa0, b0, s);
      s = wmma_f16(qa1, b1, s);
      int kcol = kc + n * 16 + lm;
#pragma unroll
      for (int r = 0; r < 8; ++r) {
        int qrow = q0 + wv * 16 + r + 8 * half;
        unsigned char mb = msk[((size_t)b * SS + qrow) * SS + kcol];
        s[r] = mb ? -1.0e7f : s[r];
      }
      st[n] = s;
    }

    // online softmax (row reductions across 16-lane N-groups)
#pragma unroll
    for (int r = 0; r < 8; ++r) {
      float mc = fmaxf(fmaxf(st[0][r], st[1][r]), fmaxf(st[2][r], st[3][r]));
      mc = fmaxf(mc, __shfl_xor(mc, 1, 32));
      mc = fmaxf(mc, __shfl_xor(mc, 2, 32));
      mc = fmaxf(mc, __shfl_xor(mc, 4, 32));
      mc = fmaxf(mc, __shfl_xor(mc, 8, 32));
      float newm  = fmaxf(mrow[r], mc);
      float alpha = __expf(mrow[r] - newm);
      mrow[r] = newm;
      float rs = 0.0f;
#pragma unroll
      for (int n = 0; n < 4; ++n) {
        float p = __expf(st[n][r] - newm);
        st[n][r] = p;
        rs += p;
      }
      rs += __shfl_xor(rs, 1, 32);
      rs += __shfl_xor(rs, 2, 32);
      rs += __shfl_xor(rs, 4, 32);
      rs += __shfl_xor(rs, 8, 32);
      lrow[r] = lrow[r] * alpha + rs;
#pragma unroll
      for (int nd = 0; nd < 4; ++nd) o[nd][r] *= alpha;
    }

    // P: C-layout f32 -> A-layout f16 via wave-private LDS staging
    _Float16* pw = &PL[wv * 16 * KVSTR];
#pragma unroll
    for (int n = 0; n < 4; ++n)
#pragma unroll
      for (int r = 0; r < 8; ++r)
        pw[(r + 8 * half) * KVSTR + n * 16 + lm] = (_Float16)st[n][r];

    // O += P x V  (2 k-chunks of 32 x 4 d-tiles)
#pragma unroll
    for (int c = 0; c < 2; ++c) {
      const _Float16* pr = pw + lm * KVSTR;
      h16 pa = cat8(*(const h8*)(pr + 32 * c + 16 * half),
                    *(const h8*)(pr + 32 * c + 16 * half + 8));
#pragma unroll
      for (int nd = 0; nd < 4; ++nd) {
        const _Float16* vr = &VL[(nd * 16 + lm) * KVSTR + 32 * c + 16 * half];
        h16 vb = cat8(*(const h8*)vr, *(const h8*)(vr + 8));
        o[nd] = wmma_f16(pa, vb, o[nd]);
      }
    }
    __syncthreads();
  }

  // normalize and write out [B,S,D] fp32
#pragma unroll
  for (int r = 0; r < 8; ++r) {
    int qrow  = q0 + wv * 16 + r + 8 * half;
    float inv = 1.0f / lrow[r];
    size_t base = ((size_t)b * SS + qrow) * DD + h * DKK;
#pragma unroll
    for (int nd = 0; nd < 4; ++nd)
      Out[base + nd * 16 + lm] = o[nd][r] * inv;
  }
}

// -------------------------------------------------------------------------
extern "C" void kernel_launch(void* const* d_in, const int* in_sizes, int n_in,
                              void* d_out, int out_size, void* d_ws, size_t ws_size,
                              hipStream_t stream) {
  (void)in_sizes; (void)n_in; (void)out_size; (void)ws_size;
  const float* q  = (const float*)d_in[0];
  const float* k  = (const float*)d_in[1];
  const float* v  = (const float*)d_in[2];
  const unsigned char* msk = (const unsigned char*)d_in[3];
  const float* Wq = (const float*)d_in[4];
  const float* bq = (const float*)d_in[5];
  const float* Wk = (const float*)d_in[6];
  const float* bk = (const float*)d_in[7];
  const float* Wv = (const float*)d_in[8];
  const float* bv = (const float*)d_in[9];
  float* out = (float*)d_out;

  const int NX = BB * SS * DD;   // 4M elements
  const int NW = DD * DD;        // 1M elements
  const int NH = BB * HH * SS * DKK; // 4M elements

  _Float16* X16q = (_Float16*)d_ws;
  _Float16* X16k = X16q + NX;
  _Float16* X16v = X16k + NX;
  _Float16* W16q = X16v + NX;
  _Float16* W16k = W16q + NW;
  _Float16* W16v = W16k + NW;
  _Float16* Qh16 = W16v + NW;
  _Float16* Kh16 = Qh16 + NH;
  _Float16* Vt16 = Kh16 + NH;    // total ~54 MB of workspace

  f32_to_f16_kernel<<<4096, 256, 0, stream>>>(q,  X16q, NX);
  f32_to_f16_kernel<<<4096, 256, 0, stream>>>(k,  X16k, NX);
  f32_to_f16_kernel<<<4096, 256, 0, stream>>>(v,  X16v, NX);
  f32_to_f16_kernel<<<1024, 256, 0, stream>>>(Wq, W16q, NW);
  f32_to_f16_kernel<<<1024, 256, 0, stream>>>(Wk, W16k, NW);
  f32_to_f16_kernel<<<1024, 256, 0, stream>>>(Wv, W16v, NW);

  dim3 pg(BB * SS / 128, DD / 64, 1);   // (32, 16)
  proj_kernel<<<pg, 256, 0, stream>>>(X16q, W16q, bq, Qh16, 0);
  proj_kernel<<<pg, 256, 0, stream>>>(X16k, W16k, bk, Kh16, 1);
  proj_kernel<<<pg, 256, 0, stream>>>(X16v, W16v, bv, Vt16, 2);

  dim3 ag(SS / 64, HH, BB);             // (32, 16, 2)
  attn_kernel<<<ag, 128, 0, stream>>>(Qh16, Kh16, Vt16, msk, out);
}
